// TaylorSeriesLinearAttn_58746562675086
// MI455X (gfx1250) — compile-verified, matrix operand order checked by reading, not run on hardware
//
#include <hip/hip_runtime.h>
#include <hip/hip_bf16.h>

typedef __attribute__((ext_vector_type(16))) __bf16 v16bf;
typedef __attribute__((ext_vector_type(8)))  float  v8f;

#define B_SZ   4
#define N_SZ   8192
#define DIMM   1024
#define NHEADS 8
#define DH     16
#define DIN    128                 // NHEADS*DH
#define NT     (B_SZ * N_SZ)       // 32768 tokens
#define DEXP   273                 // 1 + 16 + 256
#define DPAD   288                 // padded to 18 * 16
#define SQRT_HALF 0.70710678118654752f

// ---------------------------------------------------------------------------
// WMMA 16x16x32 bf16 fragment layout (ISA 7.12.2, wave32):
//   lane L owns (m or n) = L&15 in half = L>>4; its 16 bf16 elements cover
//   K = {half*8 .. half*8+7} U {16+half*8 .. 16+half*8+7}.
// Tiles staged in *fragment order*: tile[(frag_tile*32 + lane)*16 + slot], so a
// fragment load is one contiguous 32B read per lane (ds/global_load_b128 x2).
// Inverse map for element (mn, k):
// ---------------------------------------------------------------------------
__device__ __forceinline__ int frag_lane(int mn, int k) {
  return mn + 16 * ((k >> 3) & 1);
}
__device__ __forceinline__ int frag_slot(int k) {
  return 2 * ((k & 7) >> 1) + (k & 1) + ((k >= 16) ? 8 : 0);
}

// ---------------------------------------------------------------------------
// Kernel 0: one-shot weight pre-swizzle, fp32 -> bf16 fragment order.
//  wcat : [kcb 32][nt 24][lane 32][slot 16]   <- [w_q*0.25 | w_kv] (1024x384)
//  wout : [kcb 4][nt 64][lane 32][slot 16]    <- w_out (128x1024)
// Read once, then all GEMM B-fragments are direct global_load_b128 from L2.
// ---------------------------------------------------------------------------
__global__ void k_prep_w(const float* __restrict__ w_q, const float* __restrict__ w_kv,
                         const float* __restrict__ w_out,
                         __bf16* __restrict__ wcat, __bf16* __restrict__ wout) {
  const int i = blockIdx.x * blockDim.x + threadIdx.x;
  if (i < DIMM * 384) {
    const int kg = i / 384, c = i - kg * 384;
    const float val = (c < 128) ? w_q[(size_t)kg * 128 + c] * 0.25f   // fold dh^-0.5
                                : w_kv[(size_t)kg * 256 + (c - 128)];
    const int kcb = kg >> 5, k = kg & 31, nt = c >> 4, n = c & 15;
    wcat[(((size_t)kcb * 24 + nt) * 32 + frag_lane(n, k)) * 16 + frag_slot(k)] = (__bf16)val;
  }
  const int j = i - DIMM * 384;
  if (j >= 0 && j < DIN * DIMM) {
    const int kg = j >> 10, c = j & 1023;
    const int kcb = kg >> 5, k = kg & 31, nt = c >> 4, n = c & 15;
    wout[(((size_t)kcb * 64 + nt) * 32 + frag_lane(n, k)) * 16 + frag_slot(k)] =
        (__bf16)w_out[(size_t)kg * DIMM + c];
  }
}

// ---------------------------------------------------------------------------
// Kernel 1: fused RMS-norm + Q/KV projection.
// 64 tokens x 384 cols; A (normed x, bf16) staged via LDS in frag order with
// register double-buffering of the next K-step's x values (global loads fly
// under the WMMA chain); B fragments direct from pre-swizzled global wcat.
// 8 waves: 4(M) x 2(N-groups of 12 tiles); 12 acc tiles per wave.
// ---------------------------------------------------------------------------
__global__ __launch_bounds__(256) void k_norm_proj(
    const float* __restrict__ x, const float* __restrict__ gamma,
    const __bf16* __restrict__ wcat,
    float* __restrict__ qb, float* __restrict__ kbuf, float* __restrict__ vbuf) {
  __shared__ __align__(32) __bf16 As[4 * 32 * 16];    // 4 m-tiles, frag order
  __shared__ float  sscale[64];
  __shared__ float  spart[256];

  const int tid    = threadIdx.x;
  const int token0 = blockIdx.x * 64;

  // ---- row sums of squares: 4 threads per row, float4 loads ----
  {
    const int row = tid >> 2, qr = tid & 3;
    const float* xr = x + (size_t)(token0 + row) * DIMM + qr * 256;
    float s = 0.f;
    for (int c = 0; c < 256; c += 4) {
      float4 v4 = *reinterpret_cast<const float4*>(xr + c);
      s += v4.x * v4.x + v4.y * v4.y + v4.z * v4.z + v4.w * v4.w;
    }
    spart[tid] = s;
  }
  __syncthreads();
  if (tid < 64) {
    float n2 = spart[tid * 4] + spart[tid * 4 + 1] + spart[tid * 4 + 2] + spart[tid * 4 + 3];
    sscale[tid] = 32.0f / fmaxf(sqrtf(n2), 1e-12f);  // sqrt(1024)/||x||
  }

  const int wave = tid >> 5, lane = tid & 31;
  const int mtile = wave & 3, ngrp = wave >> 2;

  // staging geometry: idx = tid + u*256 -> k = tid&31 (loop-invariant),
  // row r = (tid>>5) + u*8
  const int kk = tid & 31, r0 = tid >> 5;
  const int slot = frag_slot(kk);
  const int laneadd = 16 * ((kk >> 3) & 1);
  const float* xp = x + (size_t)(token0 + r0) * DIMM + kk;

  v8f acc[12];
#pragma unroll
  for (int j = 0; j < 12; ++j)
#pragma unroll
    for (int r = 0; r < 8; ++r) acc[j][r] = 0.f;

  // preload K-step 0
  float xv[8];
  float gv = gamma[kk];
#pragma unroll
  for (int u = 0; u < 8; ++u) xv[u] = xp[(size_t)u * 8 * DIMM];

  for (int kc = 0; kc < DIMM; kc += 32) {
    __syncthreads();
    // scatter prefetched A values to frag-ordered LDS (convert to bf16)
#pragma unroll
    for (int u = 0; u < 8; ++u) {
      const int r = r0 + u * 8;
      As[(((r >> 4) * 32) + (r & 15) + laneadd) * 16 + slot] =
          (__bf16)(xv[u] * sscale[r] * gv);
    }
    __syncthreads();
    // issue loads for next K-step; they overlap the WMMA chain below
    if (kc + 32 < DIMM) {
      gv = gamma[kc + 32 + kk];
#pragma unroll
      for (int u = 0; u < 8; ++u) xv[u] = xp[(size_t)u * 8 * DIMM + kc + 32];
    }

    const v16bf a = *reinterpret_cast<const v16bf*>(As + (mtile * 32 + lane) * 16);
    const __bf16* wc = wcat + ((size_t)(kc >> 5) * 24) * 512;   // 512 = 32*16
#pragma unroll
    for (int j = 0; j < 12; ++j) {
      const v16bf bf = *reinterpret_cast<const v16bf*>(
          wc + ((ngrp * 12 + j) * 32 + lane) * 16);
      acc[j] = __builtin_amdgcn_wmma_f32_16x16x32_bf16(
          false, a, false, bf, (short)0, acc[j], false, false);
    }
  }

  // ---- store: split concat cols into q / k / v buffers ----
  const int cn = lane & 15, mh = lane >> 4;
#pragma unroll
  for (int j = 0; j < 12; ++j) {
    const int col = ngrp * 192 + j * 16 + cn;
#pragma unroll
    for (int r = 0; r < 8; ++r) {
      const int m = mtile * 16 + mh * 8 + r;
      const size_t tok = (size_t)(token0 + m);
      const float v = acc[j][r];
      if (col < 128)      qb[tok * DIN + col]           = v;
      else if (col < 256) kbuf[tok * DIN + (col - 128)] = v;
      else                vbuf[tok * DIN + (col - 256)] = v;
    }
  }
}

// ---------------------------------------------------------------------------
// Zero-fill scratch (graph-capture-safe).
// ---------------------------------------------------------------------------
__global__ void k_zero(float* __restrict__ p, int n) {
  const int i = blockIdx.x * blockDim.x + threadIdx.x;
  if (i < n) p[i] = 0.f;
}

// ---------------------------------------------------------------------------
// Kernel 2: KV state reduction. One thread per expanded feature d (virtual
// Taylor expansion of k), 128-token chunks in LDS, fp32 atomics to global.
// kvstate: [bh][DPAD][16], ksum: [bh][DPAD]
// ---------------------------------------------------------------------------
__global__ __launch_bounds__(288) void k_kv_state(
    const float* __restrict__ kbuf, const float* __restrict__ vbuf,
    float* __restrict__ kvstate, float* __restrict__ ksum) {
  __shared__ float kl[128 * 16];
  __shared__ float vl[128 * 16];
  const int chunk = blockIdx.x, bh = blockIdx.y;
  const int b = bh >> 3, h = bh & 7;
  const int tid = threadIdx.x;

  for (int idx = tid; idx < 128 * 16; idx += 288) {
    const int i = idx >> 4, e = idx & 15;
    const size_t tok = (size_t)b * N_SZ + (size_t)chunk * 128 + i;
    kl[idx] = kbuf[tok * DIN + h * DH + e];
    vl[idx] = vbuf[tok * DIN + h * DH + e];
  }
  __syncthreads();

  const int d = tid;
  if (d < DEXP) {
    float acc[16];
#pragma unroll
    for (int e = 0; e < 16; ++e) acc[e] = 0.f;
    float ks = 0.f;
    for (int i = 0; i < 128; ++i) {
      float c;
      if (d == 0)       c = 1.f;
      else if (d <= 16) c = kl[i * 16 + (d - 1)];
      else {
        const int p = d - 17;
        c = SQRT_HALF * kl[i * 16 + (p >> 4)] * kl[i * 16 + (p & 15)];
      }
      ks += c;
#pragma unroll
      for (int e = 0; e < 16; ++e) acc[e] += c * vl[i * 16 + e];
    }
    float* dst = kvstate + ((size_t)bh * DPAD + d) * 16;
#pragma unroll
    for (int e = 0; e < 16; ++e) atomicAdd(&dst[e], acc[e]);
    atomicAdd(&ksum[(size_t)bh * DPAD + d], ks);
  }
}

// ---------------------------------------------------------------------------
// Kernel 3: apply q_exp @ kv_state with 1/max(den,eps).
// Emits attn_out in bf16 *A-fragment order* so the output projection can feed
// WMMA straight from global memory:
//   attnf[(mt*4 + kcb)*512 + frag_lane(m,k)*16 + frag_slot(k)]
//   with mt = token>>4, m = token&15, col = h*16+e, kcb = col>>5, k = col&31.
// ---------------------------------------------------------------------------
__global__ __launch_bounds__(256) void k_attn_apply(
    const float* __restrict__ qb, const float* __restrict__ kvstate,
    const float* __restrict__ ksum, __bf16* __restrict__ attnf) {
  __shared__ float kvs[DEXP * 16];
  __shared__ float kss[DEXP];
  const int chunk = blockIdx.x, bh = blockIdx.y;
  const int b = bh >> 3, h = bh & 7;
  const int tid = threadIdx.x;

  const float* kvsrc = kvstate + (size_t)bh * DPAD * 16;  // first DEXP*16 contiguous
  for (int idx = tid; idx < DEXP * 16; idx += 256) kvs[idx] = kvsrc[idx];
  for (int idx = tid; idx < DEXP; idx += 256) kss[idx] = ksum[(size_t)bh * DPAD + idx];
  __syncthreads();

  const int e = tid & 15;
  const int col = h * DH + e;
  const int kcb = col >> 5, kk = col & 31;
  for (int tl = (tid >> 4); tl < 128; tl += 16) {
    const int token = b * N_SZ + chunk * 128 + tl;
    const size_t tok = (size_t)token;
    float q[16];
#pragma unroll
    for (int i = 0; i < 16; ++i) q[i] = qb[tok * DIN + h * DH + i];

    float acc = kvs[e];     // d = 0 (constant feature)
    float den = kss[0];
#pragma unroll
    for (int i = 0; i < 16; ++i) {
      acc += q[i] * kvs[(1 + i) * 16 + e];
      den += q[i] * kss[1 + i];
    }
    for (int i = 0; i < 16; ++i) {
      const float qi = SQRT_HALF * q[i];
#pragma unroll
      for (int j = 0; j < 16; ++j) {
        const float qq = qi * q[j];
        const int d = 17 + 16 * i + j;
        acc += qq * kvs[d * 16 + e];
        den += qq * kss[d];
      }
    }
    const float o = acc / fmaxf(den, 1e-5f);
    const int mt = token >> 4, m = token & 15;
    attnf[(((size_t)mt * 4 + kcb) * 32 + frag_lane(m, kk)) * 16 + frag_slot(kk)] = (__bf16)o;
  }
}

// ---------------------------------------------------------------------------
// Kernel 4: output projection [NT x 128] @ [128 x 1024] via WMMA bf16.
// Both operands pre-swizzled to fragment order in global memory: the whole
// kernel is global_load_b128 pairs -> v_wmma, no LDS, no barriers.
// Tile 64 tokens x 256 cols; 8 waves = 4M x 2N; 8 acc tiles per wave.
// ---------------------------------------------------------------------------
__global__ __launch_bounds__(256) void k_out_proj(
    const __bf16* __restrict__ attnf, const __bf16* __restrict__ woutf,
    float* __restrict__ out) {
  const int tid = threadIdx.x;
  const int m0 = blockIdx.x * 64;
  const int n0 = blockIdx.y * 256;
  const int wave = tid >> 5, lane = tid & 31;
  const int mtile = wave & 3, ngrp = wave >> 2;
  const int mt = (m0 >> 4) + mtile;   // global 16-token tile id

  v8f acc[8];
#pragma unroll
  for (int j = 0; j < 8; ++j)
#pragma unroll
    for (int r = 0; r < 8; ++r) acc[j][r] = 0.f;

#pragma unroll
  for (int kcb = 0; kcb < 4; ++kcb) {
    const v16bf a = *reinterpret_cast<const v16bf*>(
        attnf + (((size_t)mt * 4 + kcb) * 32 + lane) * 16);
#pragma unroll
    for (int j = 0; j < 8; ++j) {
      const int nt = (n0 >> 4) + ngrp * 8 + j;
      const v16bf bf = *reinterpret_cast<const v16bf*>(
          woutf + (((size_t)kcb * 64 + nt) * 32 + lane) * 16);
      acc[j] = __builtin_amdgcn_wmma_f32_16x16x32_bf16(
          false, a, false, bf, (short)0, acc[j], false, false);
    }
  }

  const int cn = lane & 15, mh = lane >> 4;
#pragma unroll
  for (int j = 0; j < 8; ++j) {
    const int col = n0 + ngrp * 128 + j * 16 + cn;
#pragma unroll
    for (int r = 0; r < 8; ++r) {
      const int m = mtile * 16 + mh * 8 + r;
      out[(size_t)(m0 + m) * DIMM + col] = acc[j][r];
    }
  }
}

// ---------------------------------------------------------------------------
extern "C" void kernel_launch(void* const* d_in, const int* in_sizes, int n_in,
                              void* d_out, int out_size, void* d_ws, size_t ws_size,
                              hipStream_t stream) {
  const float* x     = (const float*)d_in[0];
  const float* gamma = (const float*)d_in[1];
  const float* w_q   = (const float*)d_in[2];
  const float* w_kv  = (const float*)d_in[3];
  const float* w_out = (const float*)d_in[4];
  float* out = (float*)d_out;

  // workspace layout: q | k | v | kvstate | ksum (fp32) | attnf | wcat | wout (bf16)
  float* qb      = (float*)d_ws;
  float* kbuf    = qb + (size_t)NT * DIN;
  float* vbuf    = kbuf + (size_t)NT * DIN;
  float* kvstate = vbuf + (size_t)NT * DIN;
  float* ksum    = kvstate + (size_t)(B_SZ * NHEADS) * DPAD * 16;
  __bf16* attnf  = (__bf16*)(ksum + (size_t)(B_SZ * NHEADS) * DPAD);
  __bf16* wcat   = attnf + (size_t)NT * DIN;
  __bf16* woutf  = wcat + (size_t)DIMM * 384;

  // 0) pre-swizzle weights to fragment-ordered bf16 (one pass, L2-resident after)
  const int nprep = DIMM * 384 + DIN * DIMM;
  k_prep_w<<<dim3((nprep + 255) / 256), dim3(256), 0, stream>>>(w_q, w_kv, w_out, wcat, woutf);

  // 1) fused RMS-norm + Q/KV projection (WMMA)
  k_norm_proj<<<dim3(NT / 64), dim3(256), 0, stream>>>(x, gamma, wcat, qb, kbuf, vbuf);

  // 2) zero kv_state + k_sum scratch, then reduce over n
  const int nz = (B_SZ * NHEADS) * DPAD * 16 + (B_SZ * NHEADS) * DPAD;
  k_zero<<<dim3((nz + 255) / 256), dim3(256), 0, stream>>>(kvstate, nz);
  k_kv_state<<<dim3(N_SZ / 128, B_SZ * NHEADS), dim3(288), 0, stream>>>(kbuf, vbuf, kvstate, ksum);

  // 3) apply attention state per token -> fragment-ordered bf16 attn
  k_attn_apply<<<dim3(N_SZ / 128, B_SZ * NHEADS), dim3(256), 0, stream>>>(qb, kvstate, ksum, attnf);

  // 4) output projection (WMMA, no LDS)
  k_out_proj<<<dim3(NT / 64, DIMM / 256), dim3(256), 0, stream>>>(attnf, woutf, out);
}
